// SelfAttention_33097017983676
// MI455X (gfx1250) — matrix operand order repaired and checked by reading, not compiled, not run on hardware
//
#include <hip/hip_runtime.h>

// ---------------------------------------------------------------------------
// SelfAttention on CDNA5 (gfx1250), wave32 + v_wmma_f32_16x16x32_bf16
//   B=16, T=2048, D=512, A=128
//   out = [ w (B*T f32) | w_att (B*T*T f32) ]
//
// Design (from MI455X roofline):
//  - w_att write (268 MB) dominates -> single-pass scores: 16x2048 f32 strip
//    lives entirely in LDS (CDNA5: 320 KB/WGP), softmax in place, NT stream-out.
//  - Q/K/V (~16.5 MB) stay L2-resident across all attention blocks; NT hints
//    on the streaming output keep them from being displaced.
//  - All matmuls on v_wmma_f32_16x16x32_bf16 (f32 accumulate).
// ---------------------------------------------------------------------------

#define Bsz   16
#define Tsz   2048
#define Dsz   512
#define Asz   128
#define SCALE 0.08838834764831845f   // 128^-0.5
#define EPSn  1e-6f
#define SROW  2052                   // 2048 + 4 pad (bank-conflict avoidance)

typedef __attribute__((ext_vector_type(16))) __bf16 v16bf;
typedef __attribute__((ext_vector_type(8)))  float  v8f;

union U16bf { unsigned int u[8]; v16bf v; };

static __device__ __forceinline__ unsigned short f2bf(float f) {
    union { __bf16 h; unsigned short u; } cv;
    cv.h = (__bf16)f;
    return cv.u;
}

// ---------------------------------------------------------------------------
// Kernel 1: per-row mean/std (ddof=1) normalization; writes xn (bf16) and
// V = xn . Wv + bv (f32).  One wave32 per row, 8 waves per block.
// x is read exactly once -> non-temporal loads.
// ---------------------------------------------------------------------------
__global__ __launch_bounds__(256) void
sa_norm_kernel(const float* __restrict__ x, const float* __restrict__ Wv,
               const float* __restrict__ bv,
               unsigned short* __restrict__ xn_out, float* __restrict__ v_out)
{
    const int wave = threadIdx.x >> 5;
    const int lane = threadIdx.x & 31;
    const long row = (long)blockIdx.x * 8 + wave;     // < B*T = 32768

    const float* xr = x + row * Dsz;
    float vals[16];
    float s = 0.f, ss = 0.f;
#pragma unroll
    for (int i = 0; i < 16; ++i) {
        float v = __builtin_nontemporal_load(xr + lane + i * 32);
        vals[i] = v; s += v; ss += v * v;
    }
#pragma unroll
    for (int m = 16; m; m >>= 1) {
        s  += __shfl_xor(s,  m, 32);
        ss += __shfl_xor(ss, m, 32);
    }
    const float mu  = s * (1.f / (float)Dsz);
    const float var = (ss - (float)Dsz * mu * mu) * (1.f / (float)(Dsz - 1));
    const float sd  = sqrtf(fmaxf(var, 0.f));
    const float inv = 1.f / (sd + EPSn);

    float vacc = 0.f;
#pragma unroll
    for (int i = 0; i < 16; ++i) {
        const float xnv = (vals[i] - mu) * inv;
        vacc += xnv * Wv[lane + i * 32];
        xn_out[row * Dsz + lane + i * 32] = f2bf(xnv);
    }
#pragma unroll
    for (int m = 16; m; m >>= 1) vacc += __shfl_xor(vacc, m, 32);
    if (lane == 0) v_out[row] = vacc + bv[0];
}

// ---------------------------------------------------------------------------
// Kernel 2: Wq,Wk [D][A] f32 -> transposed bf16 [A][D] so the GEMM B-fragment
// k-pairs are contiguous dword loads.
// ---------------------------------------------------------------------------
__global__ __launch_bounds__(256) void
sa_wt_kernel(const float* __restrict__ Wq, const float* __restrict__ Wk,
             unsigned short* __restrict__ WqT, unsigned short* __restrict__ WkT)
{
    const int idx = blockIdx.x * 256 + threadIdx.x;   // a*512 + d, < 65536
    const int a = idx >> 9;
    const int d = idx & (Dsz - 1);
    WqT[idx] = f2bf(Wq[d * Asz + a]);
    WkT[idx] = f2bf(Wk[d * Asz + a]);
}

// ---------------------------------------------------------------------------
// Kernel 3: Q = xn @ Wq + bq, K = xn @ Wk + bk   (bf16 WMMA, f32 accum)
// grid.x = (B*T)/16 row tiles, 8 waves: wave w -> column tile n0 = w*16 for
// both Q and K (A-fragment shared per k-chunk).
// ---------------------------------------------------------------------------
__global__ __launch_bounds__(256) void
sa_proj_kernel(const unsigned short* __restrict__ xn,
               const unsigned short* __restrict__ WqT,
               const unsigned short* __restrict__ WkT,
               const float* __restrict__ bq, const float* __restrict__ bk,
               unsigned short* __restrict__ Qo, unsigned short* __restrict__ Ko)
{
    const int wave = threadIdx.x >> 5;
    const int lane = threadIdx.x & 31;
    const int M    = lane & 15;          // A-matrix row
    const int half = lane >> 4;
    const int N    = lane & 15;          // B/D column
    const long row0 = (long)blockIdx.x * 16;
    const int  n0   = wave * 16;

    v8f accQ = {}, accK = {};
#pragma unroll 4
    for (int kk = 0; kk < Dsz / 32; ++kk) {
        const int k0 = kk * 32;
        U16bf ua, ubq, ubk;
#pragma unroll
        for (int v = 0; v < 8; ++v) {
            // A (16x32 bf16) layout: lanes 0-15 K={0..7,16..23}, lanes 16-31 +8
            const int kb = ((v < 4) ? 0 : 16) + half * 8 + (v & 3) * 2;
            ua.u[v] = *(const unsigned int*)(xn + (row0 + M) * Dsz + k0 + kb);
            // B (32x16 bf16): lane N, K = same order as A (pairs contiguous in WT)
            const int kr = k0 + kb;
            ubq.u[v] = *(const unsigned int*)(WqT + (long)(n0 + N) * Dsz + kr);
            ubk.u[v] = *(const unsigned int*)(WkT + (long)(n0 + N) * Dsz + kr);
        }
        accQ = __builtin_amdgcn_wmma_f32_16x16x32_bf16(false, ua.v, false, ubq.v,
                                                       (short)0, accQ, false, false);
        accK = __builtin_amdgcn_wmma_f32_16x16x32_bf16(false, ua.v, false, ubk.v,
                                                       (short)0, accK, false, false);
    }
    const float biasq = bq[n0 + N];
    const float biask = bk[n0 + N];
#pragma unroll
    for (int r = 0; r < 8; ++r) {
        const long orow = row0 + r + 8 * half;   // D row M = r + 8*half
        Qo[orow * Asz + n0 + N] = f2bf(accQ[r] + biasq);
        Ko[orow * Asz + n0 + N] = f2bf(accK[r] + biask);
    }
}

// ---------------------------------------------------------------------------
// Kernel 4: per (b, 16-row tile): scores -> LDS strip (16 x 2052 f32),
// softmax in LDS, non-temporal stream of normalized w_att to HBM, fused
// w = softmax . V.  grid = (T/16, B), 8 waves, 131328 B dynamic LDS.
// ---------------------------------------------------------------------------
__global__ __launch_bounds__(256) void
sa_attn_kernel(const unsigned short* __restrict__ Qb,
               const unsigned short* __restrict__ Kb,
               const float* __restrict__ Vf,
               float* __restrict__ w_out, float* __restrict__ watt_out)
{
    extern __shared__ float sc[];                 // [16][SROW]
    const int wave = threadIdx.x >> 5;
    const int lane = threadIdx.x & 31;
    const int M    = lane & 15;
    const int half = lane >> 4;
    const int N    = lane & 15;
    const int b    = blockIdx.y;
    const int t0   = blockIdx.x * 16;
    const long qrow0 = (long)b * Tsz + t0;

    // Preload Q tile A-fragments (4 chunks of 32 along A-dim), kept in VGPRs.
    U16bf ua[4];
#pragma unroll
    for (int s = 0; s < 4; ++s)
#pragma unroll
        for (int v = 0; v < 8; ++v) {
            const int kb = ((v < 4) ? 0 : 16) + half * 8 + (v & 3) * 2;
            ua[s].u[v] = *(const unsigned int*)(Qb + (qrow0 + M) * Asz + s * 32 + kb);
        }

    // Score strip: each wave covers column tiles ct = wave, wave+8, ...
    for (int ct = wave; ct < Tsz / 16; ct += 8) {
        v8f acc = {};
        const long krow = ((long)b * Tsz + ct * 16 + N) * Asz;
#pragma unroll
        for (int s = 0; s < 4; ++s) {
            U16bf ub;
#pragma unroll
            for (int v = 0; v < 8; ++v)
                ub.u[v] = *(const unsigned int*)(Kb + krow + s * 32 + ((v < 4) ? 0 : 16) + half * 8 + (v & 3) * 2);
            acc = __builtin_amdgcn_wmma_f32_16x16x32_bf16(false, ua[s].v, false, ub.v,
                                                          (short)0, acc, false, false);
        }
#pragma unroll
        for (int r = 0; r < 8; ++r)
            sc[(r + 8 * half) * SROW + ct * 16 + N] = acc[r] * SCALE;
    }
    __syncthreads();

    // Softmax + w: one wave per 2 rows, wave32 shfl reductions.
    const float* Vb = Vf + (long)b * Tsz;
    for (int rr = 0; rr < 2; ++rr) {
        const int row = wave * 2 + rr;
        float* srow = sc + row * SROW;

        float m = -3.4e38f;
        for (int c = lane; c < Tsz; c += 32) m = fmaxf(m, srow[c]);
#pragma unroll
        for (int msk = 16; msk; msk >>= 1) m = fmaxf(m, __shfl_xor(m, msk, 32));

        float sum = 0.f, vacc = 0.f;
        for (int c = lane; c < Tsz; c += 32) {
            const float e = __expf(srow[c] - m);
            srow[c] = e;
            sum  += e;
            vacc += e * Vb[c];
        }
#pragma unroll
        for (int msk = 16; msk; msk >>= 1) {
            sum  += __shfl_xor(sum,  msk, 32);
            vacc += __shfl_xor(vacc, msk, 32);
        }
        const float inv = 1.f / sum;
        const long  obase = ((long)b * Tsz + t0 + row) * (long)Tsz;
        // Streaming, never re-read: non-temporal stores keep Q/K/V in L2.
        for (int c = lane; c < Tsz; c += 32)
            __builtin_nontemporal_store(srow[c] * inv, watt_out + obase + c);
        if (lane == 0) w_out[(long)b * Tsz + t0 + row] = vacc * inv;
    }
}

// ---------------------------------------------------------------------------
extern "C" void kernel_launch(void* const* d_in, const int* in_sizes, int n_in,
                              void* d_out, int out_size, void* d_ws, size_t ws_size,
                              hipStream_t stream)
{
    (void)in_sizes; (void)n_in; (void)out_size; (void)ws_size;

    const float* x  = (const float*)d_in[0];
    const float* Wq = (const float*)d_in[1];
    const float* bq = (const float*)d_in[2];
    const float* Wk = (const float*)d_in[3];
    const float* bk = (const float*)d_in[4];
    const float* Wv = (const float*)d_in[5];
    const float* bv = (const float*)d_in[6];

    float* w_out    = (float*)d_out;                       // B*T
    float* watt_out = (float*)d_out + (long)Bsz * Tsz;     // B*T*T

    // Workspace layout (bytes):
    char* ws = (char*)d_ws;
    const size_t NROWS = (size_t)Bsz * Tsz;                // 32768
    unsigned short* xn_b = (unsigned short*)(ws);                          // 32 MB
    unsigned short* WqT  = (unsigned short*)(ws + NROWS * Dsz * 2);        // 128 KB
    unsigned short* WkT  = WqT + (size_t)Asz * Dsz;                        // 128 KB
    unsigned short* Qb   = WkT + (size_t)Asz * Dsz;                        // 8 MB
    unsigned short* Kb   = Qb  + NROWS * Asz;                              // 8 MB
    float*          Vf   = (float*)(Kb + NROWS * Asz);                     // 128 KB

    // 1) row norm + V
    sa_norm_kernel<<<dim3(NROWS / 8), dim3(256), 0, stream>>>(x, Wv, bv, xn_b, Vf);

    // 2) transpose+convert weights to bf16
    sa_wt_kernel<<<dim3((Asz * Dsz) / 256), dim3(256), 0, stream>>>(Wq, Wk, WqT, WkT);

    // 3) Q/K projections (WMMA bf16)
    sa_proj_kernel<<<dim3(NROWS / 16), dim3(256), 0, stream>>>(xn_b, WqT, WkT, bq, bk, Qb, Kb);

    // 4) attention: scores (WMMA) -> LDS softmax -> w_att (NT) + w
    const int shmem = 16 * SROW * (int)sizeof(float);      // 131328 B
    (void)hipFuncSetAttribute((const void*)sa_attn_kernel,
                              hipFuncAttributeMaxDynamicSharedMemorySize, shmem);
    sa_attn_kernel<<<dim3(Tsz / 16, Bsz), dim3(256), shmem, stream>>>(Qb, Kb, Vf, w_out, watt_out);
}
